// LocalTransformerBlock_12756052869789
// MI455X (gfx1250) — matrix operand order repaired and checked by reading, not compiled, not run on hardware
//
#include <hip/hip_runtime.h>
#include <hip/hip_bf16.h>

// ---------------------------------------------------------------------------
// LocalTransformerBlock on gfx1250 (MI455X): all GEMMs via v_wmma_f32_16x16x32_f16
// GEMM wave tile 64x64: 16 WMMA per 16 b128 loads per k-step (1.0 loads/WMMA).
// ---------------------------------------------------------------------------

typedef _Float16 h16;
typedef __attribute__((ext_vector_type(16))) _Float16 v16h;
typedef __attribute__((ext_vector_type(8)))  _Float16 v8h;
typedef __attribute__((ext_vector_type(8)))  float    v8f;

#define BB 4
#define LL 4096
#define DD 512
#define HH 8
#define DH 64
#define MLPD 2048
#define BSZ 128
#define NB 32              // L / BS
#define MTOT (BB*LL)       // 16384 rows

// ---------------- WMMA helpers (CDNA5 layouts) -----------------------------

__device__ __forceinline__ v8f wmma32(v16h a, v16h b, v8f c) {
  // D = A(16x32 f16) x B(32x16 f16) + C(16x16 f32)
  return __builtin_amdgcn_wmma_f32_16x16x32_f16(
      /*neg_a=*/false, a, /*neg_b=*/false, b,
      /*c_mod=*/(short)0, c, /*reuse_a=*/false, /*reuse_b=*/false);
}

// A fragment 16x32 f16, row-major source with leading dim ld (elements).
// Lanes 0-15: row M=lane, K=k0..k0+15
// Lanes 16-31: row M=lane-16, K=k0+16..k0+31
__device__ __forceinline__ v16h frag_a(const h16* base, int ld, int lane) {
  const h16* p = base + (size_t)(lane & 15) * ld + ((lane >> 4) << 4);
  v8h lo = *(const v8h*)(p);
  v8h hi = *(const v8h*)(p + 8);
  v16h f;
#pragma unroll
  for (int i = 0; i < 8; ++i) { f[i] = lo[i]; f[i + 8] = hi[i]; }
  return f;
}

// B fragment 32x16 f16 from B^T stored [N][K] row-major (K contiguous), ld = K.
// Lane holds column n=lane%16; lanes 0-15: K=k0..k0+15, lanes 16-31: K=k0+16..k0+31
__device__ __forceinline__ v16h frag_b(const h16* base, int ld, int lane) {
  const h16* p = base + (size_t)(lane & 15) * ld + ((lane >> 4) << 4);
  return *(const v16h*)(p);
}

// ---------------- weight pack: dst[n*K + k] = (h16)(src[k*N + n] * scale) ---

__global__ void pack_t_kernel(const float* __restrict__ src, h16* __restrict__ dst,
                              int N, int K, float scale) {
  int i = blockIdx.x * blockDim.x + threadIdx.x;
  if (i >= N * K) return;
  int n = i / K, k = i - n * K;
  dst[i] = (h16)(src[(size_t)k * N + n] * scale);
}

// ---------------- LayerNorm (rows of 512) -> f16 ---------------------------

__global__ __launch_bounds__(256) void ln_kernel(const float* __restrict__ x,
                                                 const float* __restrict__ g,
                                                 const float* __restrict__ bt,
                                                 h16* __restrict__ y) {
  __shared__ float r1[256], r2[256];
  int row = blockIdx.x, tid = threadIdx.x;
  const float* xr = x + (size_t)row * DD;
  float a = xr[tid], b = xr[tid + 256];
  r1[tid] = a + b;
  r2[tid] = a * a + b * b;
  __syncthreads();
#pragma unroll
  for (int s = 128; s > 0; s >>= 1) {
    if (tid < s) { r1[tid] += r1[tid + s]; r2[tid] += r2[tid + s]; }
    __syncthreads();
  }
  float mu = r1[0] * (1.0f / DD);
  float var = r2[0] * (1.0f / DD) - mu * mu;
  float rinv = rsqrtf(var + 1e-6f);
  h16* yr = y + (size_t)row * DD;
  yr[tid]       = (h16)((a - mu) * rinv * g[tid] + bt[tid]);
  yr[tid + 256] = (h16)((b - mu) * rinv * g[tid + 256] + bt[tid + 256]);
}

// ---------------- generic WMMA GEMM: C[M,N] = A[M,K] * Bt[N,K]^T -----------
// block tile 256x128, 8 waves (4 along M x 2 along N), wave tile 64x64.
// EPI: 0 = store f16 ; 1 = bias+gelu(tanh) f16 ; 2 = +resid -> f32 ; 3 = +bias+resid -> f32

__device__ __forceinline__ float gelu_tanh(float x) {
  float x3 = x * x * x;
  return 0.5f * x * (1.0f + tanhf(0.7978845608028654f * (x + 0.044715f * x3)));
}

template <int EPI>
__global__ __launch_bounds__(256) void gemm_kernel(
    const h16* __restrict__ A, const h16* __restrict__ Bt,
    int N, int K,
    float* __restrict__ outF, h16* __restrict__ outH,
    const float* __restrict__ bias, const float* __restrict__ resid) {
  int lane = threadIdx.x & 31;
  int wave = threadIdx.x >> 5;
  int m0 = blockIdx.y * 256 + (wave >> 1) * 64;
  int n0 = blockIdx.x * 128 + (wave & 1) * 64;

  v8f acc[4][4];
#pragma unroll
  for (int i = 0; i < 4; ++i)
#pragma unroll
    for (int j = 0; j < 4; ++j) acc[i][j] = (v8f)0.0f;

  for (int k0 = 0; k0 < K; k0 += 32) {
    v16h a[4];
#pragma unroll
    for (int mi = 0; mi < 4; ++mi)
      a[mi] = frag_a(A + (size_t)(m0 + mi * 16) * K + k0, K, lane);
#pragma unroll
    for (int j = 0; j < 4; ++j) {
      v16h b = frag_b(Bt + (size_t)(n0 + j * 16) * K + k0, K, lane);
#pragma unroll
      for (int mi = 0; mi < 4; ++mi)
        acc[mi][j] = wmma32(a[mi], b, acc[mi][j]);
    }
  }

  int rl = (lane >> 4) * 8;
  int cl = lane & 15;
#pragma unroll
  for (int mi = 0; mi < 4; ++mi) {
#pragma unroll
    for (int j = 0; j < 4; ++j) {
#pragma unroll
      for (int r = 0; r < 8; ++r) {
        int row = m0 + mi * 16 + rl + r;
        int col = n0 + j * 16 + cl;
        size_t idx = (size_t)row * N + col;
        float v = acc[mi][j][r];
        if (EPI == 0) {
          outH[idx] = (h16)v;
        } else if (EPI == 1) {
          outH[idx] = (h16)gelu_tanh(v + bias[col]);
        } else if (EPI == 2) {
          outF[idx] = v + resid[idx];
        } else {
          outF[idx] = v + bias[col] + resid[idx];
        }
      }
    }
  }
}

// ---------------- fused QKV GEMM with scatter epilogue ---------------------
// A = xin[MTOT][512] f16, Bt = wqkv_t[1536][512] f16 (q rows pre-scaled by DH^-0.5).
// q,k -> [b,blk,h][BS][DH] row-major; v -> [b,blk,h][DH][BS] (transposed).

__global__ __launch_bounds__(256) void qkv_kernel(
    const h16* __restrict__ A, const h16* __restrict__ Bt,
    h16* __restrict__ qb, h16* __restrict__ kb, h16* __restrict__ vb) {
  const int K = DD;
  int lane = threadIdx.x & 31;
  int wave = threadIdx.x >> 5;
  int m0 = blockIdx.y * 256 + (wave >> 1) * 64;
  int n0 = blockIdx.x * 128 + (wave & 1) * 64;

  v8f acc[4][4];
#pragma unroll
  for (int i = 0; i < 4; ++i)
#pragma unroll
    for (int j = 0; j < 4; ++j) acc[i][j] = (v8f)0.0f;

  for (int k0 = 0; k0 < K; k0 += 32) {
    v16h a[4];
#pragma unroll
    for (int mi = 0; mi < 4; ++mi)
      a[mi] = frag_a(A + (size_t)(m0 + mi * 16) * K + k0, K, lane);
#pragma unroll
    for (int j = 0; j < 4; ++j) {
      v16h b = frag_b(Bt + (size_t)(n0 + j * 16) * K + k0, K, lane);
#pragma unroll
      for (int mi = 0; mi < 4; ++mi)
        acc[mi][j] = wmma32(a[mi], b, acc[mi][j]);
    }
  }

  int rl = (lane >> 4) * 8;
  int cl = lane & 15;
#pragma unroll
  for (int mi = 0; mi < 4; ++mi) {
#pragma unroll
    for (int j = 0; j < 4; ++j) {
#pragma unroll
      for (int r = 0; r < 8; ++r) {
        int row = m0 + mi * 16 + rl + r;
        int col = n0 + j * 16 + cl;
        int which = col >> 9;            // 0=q 1=k 2=v
        int n = col & (DD - 1);
        int h = n >> 6, d = n & 63;
        int b = row >> 12;               // / 4096
        int l = row & 4095;
        int blk = l >> 7, rr = l & 127;
        size_t p = ((size_t)(b * NB + blk) * HH + h);
        h16 v = (h16)acc[mi][j][r];
        if (which == 0)      qb[p * (BSZ * DH) + (size_t)rr * DH + d] = v;
        else if (which == 1) kb[p * (BSZ * DH) + (size_t)rr * DH + d] = v;
        else                 vb[p * (DH * BSZ) + (size_t)d * BSZ + rr] = v;
      }
    }
  }
}

// ---------------- block-local attention ------------------------------------
// One workgroup per (b, blk, h). 8 waves; wave w owns query rows [16w,16w+16).
// scores via WMMA -> register softmax (shfl reductions) -> P f16 in LDS ->
// ctx = P x V^T via WMMA -> scatter ctx to [b*L + l][h*64 + d] f16.

__global__ __launch_bounds__(256) void attn_kernel(
    const h16* __restrict__ qb, const h16* __restrict__ kb,
    const h16* __restrict__ vb, h16* __restrict__ ctx) {
  __shared__ __align__(32) h16 sP[8][16 * BSZ];   // 32 KB

  int p = blockIdx.x;                  // (b*NB + blk)*HH + h
  int h = p & (HH - 1);
  int bblk = p >> 3;
  int b = bblk >> 5;
  int blk = bblk & (NB - 1);

  const h16* q  = qb + (size_t)p * (BSZ * DH);
  const h16* kk = kb + (size_t)p * (BSZ * DH);
  const h16* vt = vb + (size_t)p * (DH * BSZ);

  int lane = threadIdx.x & 31;
  int wave = threadIdx.x >> 5;
  int m0 = wave * 16;

  // ---- scores: 16 x 128, K = 64 ----
  v8f sc[8];
#pragma unroll
  for (int nt = 0; nt < 8; ++nt) sc[nt] = (v8f)0.0f;
#pragma unroll
  for (int k0 = 0; k0 < DH; k0 += 32) {
    v16h a = frag_a(q + (size_t)m0 * DH + k0, DH, lane);
#pragma unroll
    for (int nt = 0; nt < 8; ++nt) {
      v16h bf = frag_b(kk + (size_t)(nt * 16) * DH + k0, DH, lane);
      sc[nt] = wmma32(a, bf, sc[nt]);
    }
  }

  // ---- softmax over 128 cols, fully in registers + shuffles ----
  // element (nt, r): row = rl + r (rl = 8*(lane>=16)), col = nt*16 + (lane&15)
  float mx[8];
#pragma unroll
  for (int r = 0; r < 8; ++r) {
    float m = -1e30f;
#pragma unroll
    for (int nt = 0; nt < 8; ++nt) m = fmaxf(m, sc[nt][r]);
    m = fmaxf(m, __shfl_xor(m, 1));
    m = fmaxf(m, __shfl_xor(m, 2));
    m = fmaxf(m, __shfl_xor(m, 4));
    m = fmaxf(m, __shfl_xor(m, 8));
    mx[r] = m;
  }
#pragma unroll
  for (int nt = 0; nt < 8; ++nt)
#pragma unroll
    for (int r = 0; r < 8; ++r) sc[nt][r] = __expf(sc[nt][r] - mx[r]);
  float inv[8];
#pragma unroll
  for (int r = 0; r < 8; ++r) {
    float s = 0.0f;
#pragma unroll
    for (int nt = 0; nt < 8; ++nt) s += sc[nt][r];
    s += __shfl_xor(s, 1);
    s += __shfl_xor(s, 2);
    s += __shfl_xor(s, 4);
    s += __shfl_xor(s, 8);
    inv[r] = 1.0f / s;
  }
  int rl = (lane >> 4) * 8;
  int cl = lane & 15;
#pragma unroll
  for (int nt = 0; nt < 8; ++nt)
#pragma unroll
    for (int r = 0; r < 8; ++r)
      sP[wave][(rl + r) * BSZ + nt * 16 + cl] = (h16)(sc[nt][r] * inv[r]);
  __syncthreads();

  // ---- ctx = P(16x128) x V^T(B^T = vt[64][128]) -> 16 x 64 ----
  v8f ca[4];
#pragma unroll
  for (int j = 0; j < 4; ++j) ca[j] = (v8f)0.0f;
  const h16* Pw = &sP[wave][0];
#pragma unroll
  for (int k0 = 0; k0 < BSZ; k0 += 32) {
    v16h a = frag_a(Pw + k0, BSZ, lane);
#pragma unroll
    for (int j = 0; j < 4; ++j) {
      v16h bf = frag_b(vt + (size_t)(j * 16) * BSZ + k0, BSZ, lane);
      ca[j] = wmma32(a, bf, ca[j]);
    }
  }
#pragma unroll
  for (int j = 0; j < 4; ++j)
#pragma unroll
    for (int r = 0; r < 8; ++r) {
      int l = blk * BSZ + m0 + rl + r;
      int d = j * 16 + cl;
      ctx[((size_t)b * LL + l) * DD + h * DH + d] = (h16)ca[j][r];
    }
}

// ---------------------------------------------------------------------------
// host side
// ---------------------------------------------------------------------------

extern "C" void kernel_launch(void* const* d_in, const int* in_sizes, int n_in,
                              void* d_out, int out_size, void* d_ws, size_t ws_size,
                              hipStream_t stream) {
  const float* inputs    = (const float*)d_in[0];
  const float* ln1_scale = (const float*)d_in[1];
  const float* ln1_bias  = (const float*)d_in[2];
  const float* wq        = (const float*)d_in[3];
  const float* wk        = (const float*)d_in[4];
  const float* wv        = (const float*)d_in[5];
  const float* wo        = (const float*)d_in[6];
  const float* ln2_scale = (const float*)d_in[7];
  const float* ln2_bias  = (const float*)d_in[8];
  const float* w1        = (const float*)d_in[9];
  const float* b1        = (const float*)d_in[10];
  const float* w2        = (const float*)d_in[11];
  const float* b2        = (const float*)d_in[12];

  char* ws = (char*)d_ws;
  size_t o = 0;
  auto take = [&](size_t bytes) { size_t cur = o; o += (bytes + 255) & ~(size_t)255; return cur; };

  h16* wqkv_t = (h16*)(ws + take((size_t)3 * DD * DD * 2));        // [1536][512]
  h16* wo_t   = (h16*)(ws + take((size_t)DD * DD * 2));            // [512][512]
  h16* w1_t   = (h16*)(ws + take((size_t)MLPD * DD * 2));          // [2048][512]
  h16* w2_t   = (h16*)(ws + take((size_t)DD * MLPD * 2));          // [512][2048]
  h16* xin    = (h16*)(ws + take((size_t)MTOT * DD * 2));          // LN1 out (reused as ctx)
  h16* qbuf   = (h16*)(ws + take((size_t)MTOT * DD * 2));          // reused as y (LN2 out)
  h16* kbuf   = (h16*)(ws + take((size_t)MTOT * DD * 2));
  h16* vbuf   = (h16*)(ws + take((size_t)MTOT * DD * 2));
  float* xbuf = (float*)(ws + take((size_t)MTOT * DD * 4));        // attn residual output
  h16* hbuf   = (h16*)(ws + take((size_t)MTOT * MLPD * 2));        // MLP hidden
  h16* ctxbuf = xin;    // xin dead after QKV GEMM
  h16* ybuf   = qbuf;   // q dead after attention
  (void)ws_size; (void)in_sizes; (void)n_in; (void)out_size;

  const float qscale = 0.125f;  // DH^-0.5 folded into Wq

  // weight packing (cast + transpose to [N][K] f16)
  {
    int n = DD * DD, thr = 256;
    pack_t_kernel<<<(n + thr - 1) / thr, thr, 0, stream>>>(wq, wqkv_t,            DD, DD, qscale);
    pack_t_kernel<<<(n + thr - 1) / thr, thr, 0, stream>>>(wk, wqkv_t + DD * DD,  DD, DD, 1.0f);
    pack_t_kernel<<<(n + thr - 1) / thr, thr, 0, stream>>>(wv, wqkv_t + 2*DD*DD,  DD, DD, 1.0f);
    pack_t_kernel<<<(n + thr - 1) / thr, thr, 0, stream>>>(wo, wo_t,              DD, DD, 1.0f);
    int n1 = MLPD * DD;
    pack_t_kernel<<<(n1 + thr - 1) / thr, thr, 0, stream>>>(w1, w1_t, MLPD, DD, 1.0f);
    pack_t_kernel<<<(n1 + thr - 1) / thr, thr, 0, stream>>>(w2, w2_t, DD, MLPD, 1.0f);
  }

  // LN1 -> xin (f16)
  ln_kernel<<<MTOT, 256, 0, stream>>>(inputs, ln1_scale, ln1_bias, xin);

  // fused QKV projection (M=16384, N=1536, K=512), block tile 256x128
  qkv_kernel<<<dim3(3 * DD / 128, MTOT / 256), 256, 0, stream>>>(xin, wqkv_t, qbuf, kbuf, vbuf);

  // block-local attention: 4*32*8 = 1024 workgroups
  attn_kernel<<<BB * NB * HH, 256, 0, stream>>>(qbuf, kbuf, vbuf, ctxbuf);

  // output projection + residual: xbuf = ctx * Wo^T + inputs  (f32)
  gemm_kernel<2><<<dim3(DD / 128, MTOT / 256), 256, 0, stream>>>(
      ctxbuf, wo_t, DD, DD, xbuf, nullptr, nullptr, inputs);

  // LN2 -> y (f16)
  ln_kernel<<<MTOT, 256, 0, stream>>>(xbuf, ln2_scale, ln2_bias, ybuf);

  // MLP1: h = gelu(y * W1^T + b1)  (f16)
  gemm_kernel<1><<<dim3(MLPD / 128, MTOT / 256), 256, 0, stream>>>(
      ybuf, w1_t, MLPD, DD, nullptr, hbuf, b1, nullptr);

  // MLP2: out = h * W2^T + b2 + xbuf  (f32 -> d_out)
  gemm_kernel<3><<<dim3(DD / 128, MTOT / 256), 256, 0, stream>>>(
      hbuf, w2_t, DD, MLPD, (float*)d_out, nullptr, b2, xbuf);
}
